// HyperedgeAttention_87686052315286
// MI455X (gfx1250) — matrix operand order repaired and checked by reading, not compiled, not run on hardware
//
#include <hip/hip_runtime.h>
#include <hip/hip_bf16.h>

typedef _Float16 v16h __attribute__((ext_vector_type(16)));
typedef _Float16 h8   __attribute__((ext_vector_type(8)));
typedef _Float16 h4   __attribute__((ext_vector_type(4)));
typedef float    v8f  __attribute__((ext_vector_type(8)));

#define M_ROWS 32
#define D_DIM  128
#define O_DIM  64

// Branchless tanh: one v_exp_f32 + one v_rcp_f32, no divergence, saturates for large |x|.
__device__ __forceinline__ float fast_tanh(float x) {
    float ax = fabsf(x);
    float e  = __expf(-2.0f * ax);
    float r  = (1.0f - e) * __builtin_amdgcn_rcpf(1.0f + e);
    return copysignf(r, x);
}

// One-shot prep: W [128,64] f32 (row d, col o)  ->  W^T [64,128] f16 in d_ws.
__global__ __launch_bounds__(128) void prep_weights_kernel(
    const float* __restrict__ g_w, _Float16* __restrict__ g_wt)
{
    int i = blockIdx.x * 128 + threadIdx.x;   // 64 blocks * 128 = 8192 elements
    int d = i >> 6;
    int o = i & 63;
    g_wt[o * D_DIM + d] = (_Float16)g_w[i];   // coalesced read, scattered b16 write
}

__global__ __launch_bounds__(128) void hyperedge_attn_kernel(
    const float* __restrict__ g_in,       // [N, 32, 128]
    const _Float16* __restrict__ g_wt,    // [64, 128] f16 (W^T, contiguous K)
    const float* __restrict__ g_bias,     // [64]
    const float* __restrict__ g_aw,       // [64]
    float* __restrict__ g_out,            // [N, 128]
    float* __restrict__ g_alphas,         // [N, 32]
    int N)
{
    __shared__ float    s_in_f32[M_ROWS * D_DIM];   // 16 KB exact copy for final pass
    __shared__ _Float16 s_in_f16[M_ROWS * D_DIM];   // 8 KB  WMMA A operand
    __shared__ float    s_logits[M_ROWS];
    __shared__ float    s_alpha[M_ROWS];

    const int tid = threadIdx.x;
    const int n   = blockIdx.x;
    const float* in_n = g_in + (size_t)n * (M_ROWS * D_DIM);

    // ---- stage inputs: fp32 copy + f16 copy (b128 global loads) ----
    for (int i = tid; i < (M_ROWS * D_DIM) / 4; i += 128) {
        float4 v = ((const float4*)in_n)[i];
        ((float4*)s_in_f32)[i] = v;
        h4 hv = { (_Float16)v.x, (_Float16)v.y, (_Float16)v.z, (_Float16)v.w };
        *(h4*)&s_in_f16[4 * i] = hv;
    }
    if (tid < M_ROWS) s_logits[tid] = 0.0f;
    __syncthreads();

    // ---- per-wave tile assignment ----
    const int wave   = tid >> 5;
    const int lane   = tid & 31;
    const int lhalf  = lane >> 4;     // 0: lanes 0-15, 1: lanes 16-31
    const int l16    = lane & 15;
    const int mtile  = wave & 1;          // rows 0-15 or 16-31
    const int ntile0 = (wave >> 1) * 2;   // two adjacent 16-col tiles of O

    // A operand base: row = mtile*16 + l16; per-lane K base offset = lhalf*16
    const _Float16* aptr  = &s_in_f16[(mtile * 16 + l16) * D_DIM + lhalf * 16];
    // B operands straight from global W^T (16 KB, WGP$/L2 resident):
    // per lane 16 contiguous K halves starting at K = koff + lhalf*16
    const _Float16* bptr0 = &g_wt[(ntile0 * 16 + l16) * D_DIM + lhalf * 16];
    const _Float16* bptr1 = &g_wt[((ntile0 + 1) * 16 + l16) * D_DIM + lhalf * 16];

    v8f acc0 = {};
    v8f acc1 = {};
    #pragma unroll
    for (int ks = 0; ks < 4; ++ks) {
        const int koff = ks * 32;
        union { v16h v; h8 h[2]; } A, B0, B1;
        // A operand in this code's contiguous K order (B0/B1 match): VGPRs 0-3 = K[base..base+7], VGPRs 4-7 = K[base+8..base+15]
        A.h[0] = *(const h8*)(aptr + koff);
        A.h[1] = *(const h8*)(aptr + koff + 8);
        B0.h[0] = *(const h8*)(bptr0 + koff);
        B0.h[1] = *(const h8*)(bptr0 + koff + 8);
        B1.h[0] = *(const h8*)(bptr1 + koff);
        B1.h[1] = *(const h8*)(bptr1 + koff + 8);
        acc0 = __builtin_amdgcn_wmma_f32_16x16x32_f16(false, A.v, false, B0.v,
                                                      (short)0, acc0, false, false);
        acc1 = __builtin_amdgcn_wmma_f32_16x16x32_f16(false, A.v, false, B1.v,
                                                      (short)0, acc1, false, false);
    }

    // ---- fused bias + tanh + attn-weight dot (logits partial per lane) ----
    const int col0 = ntile0 * 16 + l16;
    const int col1 = (ntile0 + 1) * 16 + l16;
    const float b0v = g_bias[col0];
    const float b1v = g_bias[col1];
    const float a0v = g_aw[col0];
    const float a1v = g_aw[col1];
    float part[8];
    #pragma unroll
    for (int r = 0; r < 8; ++r)
        part[r] = fast_tanh(acc0[r] + b0v) * a0v + fast_tanh(acc1[r] + b1v) * a1v;

    // reduce over the 16 lanes that share each row (xor offsets stay within half)
    #pragma unroll
    for (int off = 8; off >= 1; off >>= 1) {
        #pragma unroll
        for (int r = 0; r < 8; ++r)
            part[r] += __shfl_xor(part[r], off, 32);
    }
    if (l16 == 0) {
        const int mbase = mtile * 16 + lhalf * 8;   // C/D layout: VGPR r -> M = r + lhalf*8
        #pragma unroll
        for (int r = 0; r < 8; ++r)
            atomicAdd(&s_logits[mbase + r], part[r]);
    }
    __syncthreads();

    // ---- softmax over the 32 logits (wave 0) ----
    if (tid < 32) {
        float x = s_logits[tid];
        float mx = x;
        #pragma unroll
        for (int off = 16; off >= 1; off >>= 1)
            mx = fmaxf(mx, __shfl_xor(mx, off, 32));
        float e = __expf(x - mx);
        float s = e;
        #pragma unroll
        for (int off = 16; off >= 1; off >>= 1)
            s += __shfl_xor(s, off, 32);
        float al = e * __builtin_amdgcn_rcpf(s);
        s_alpha[tid] = al;
        g_alphas[(size_t)n * M_ROWS + tid] = al;
    }
    __syncthreads();

    // ---- output[n, d] = sum_m X_f32[m, d] * alpha[m]  (thread d = tid) ----
    float accd = 0.0f;
    #pragma unroll
    for (int m = 0; m < M_ROWS; ++m)
        accd += s_in_f32[m * D_DIM + tid] * s_alpha[m];
    g_out[(size_t)n * D_DIM + tid] = accd;
}

extern "C" void kernel_launch(void* const* d_in, const int* in_sizes, int n_in,
                              void* d_out, int out_size, void* d_ws, size_t ws_size,
                              hipStream_t stream) {
    const float* g_in   = (const float*)d_in[0];
    const float* g_w    = (const float*)d_in[1];
    const float* g_bias = (const float*)d_in[2];
    const float* g_aw   = (const float*)d_in[3];
    const int N = in_sizes[0] / (M_ROWS * D_DIM);

    float* g_out    = (float*)d_out;
    float* g_alphas = g_out + (size_t)N * D_DIM;

    _Float16* g_wt = (_Float16*)d_ws;   // 64*128*2 = 16 KB scratch

    hipLaunchKernelGGL(prep_weights_kernel, dim3((D_DIM * O_DIM) / 128), dim3(128),
                       0, stream, g_w, g_wt);
    hipLaunchKernelGGL(hyperedge_attn_kernel, dim3(N), dim3(128), 0, stream,
                       g_in, g_wt, g_bias, g_aw, g_out, g_alphas, N);
}
